// CFKANLayer_4157528343266
// MI455X (gfx1250) — matrix operand order repaired and checked by reading, not compiled, not run on hardware
//
#include <hip/hip_runtime.h>
#include <hip/hip_bf16.h>

// CFKAN layer for MI455X (gfx1250, wave32, WMMA).
//   y[N,O] = sum_{d,k} T_k(tanh(x[n,d])) * C[o,d,k] + bias[o]
// == GEMM with M=65536, N=256, reduction R=D*K=2048.
// tanh(x) is staged once into LDS as f16 (natural column order; each lane's A inputs are
// one adjacent pair); the inner loop builds f16 A fragments via the Chebyshev
// recurrence, prefetches B fragments one K-step ahead (unconditionally, with a
// padded weight buffer so addresses stay induction variables), and accumulates
// in f32 via v_wmma_f32_16x16x32_f16 (8 WMMA per K-step per wave).

typedef __attribute__((ext_vector_type(16))) _Float16 v16h;
typedef __attribute__((ext_vector_type(4)))  _Float16 v4h;
typedef __attribute__((ext_vector_type(2)))  _Float16 v2h;
typedef __attribute__((ext_vector_type(8)))  float    v8f;

#define N_TOK 65536
#define DIM   256
#define OUTC  256
#define DEG   8
#define RED   (DIM * DEG)   // 2048 reduction length
#define XPADH 260           // padded LDS row stride in halfwords (130 dwords,
                            // 130 % 64 == 2 -> 16 rows hit 16 distinct banks)

// Branch-free tanh: hardware v_tanh_f32 when available, else exp2/rcp formula.
static __device__ __forceinline__ float fast_tanh(float v) {
#if __has_builtin(__builtin_amdgcn_tanhf)
    return __builtin_amdgcn_tanhf(v);
#else
    float e = __builtin_amdgcn_exp2f(v * 2.8853900817779268f);  // 2*log2(e)
    return 1.0f - 2.0f * __builtin_amdgcn_rcpf(e + 1.0f);
#endif
}

// Chebyshev T_0..T_7 of xt into 8 consecutive f16 lanes of an A fragment.
static __device__ __forceinline__ void cheby8(float xt, v16h& a, int base) {
    float tp = 1.0f, tc = xt;
    a[base + 0] = (_Float16)1.0f;
    a[base + 1] = (_Float16)xt;
    #pragma unroll
    for (int c = 2; c < 8; ++c) {
        float tn = 2.0f * xt * tc - tp;
        a[base + c] = (_Float16)tn;
        tp = tc; tc = tn;
    }
}

// ---------------------------------------------------------------------------
// Kernel 1: cheby_coeffs (O, D, K) f32 -> f16 workspace, same layout.
// Row o of the f16 buffer is that channel's 2048 weights in reduction order.
// ---------------------------------------------------------------------------
__global__ void cvt_coeffs_f16(const float* __restrict__ src,
                               _Float16* __restrict__ dst) {
    int i = blockIdx.x * blockDim.x + threadIdx.x;    // 0 .. 131071 (x4 elems)
    float4 v = ((const float4*)src)[i];
    v4h o = { (_Float16)v.x, (_Float16)v.y, (_Float16)v.z, (_Float16)v.w };
    ((v4h*)dst)[i] = o;
}

// ---------------------------------------------------------------------------
// Kernel 2: fused tanh + Chebyshev expansion + WMMA GEMM + bias.
// Block: 256 threads (8 waves), owns 64 rows x 256 cols of y.
// Wave (rg, nsub): 32 rows x 64 cols -> 8 f32 16x16 accumulators; each B
// fragment feeds 2 WMMAs (two A row-groups), halving B/L2 traffic.
// ---------------------------------------------------------------------------
__global__ __launch_bounds__(256) void kan_wmma_gemm(
    const float*    __restrict__ x,     // (65536, 256) f32
    const _Float16* __restrict__ B,     // (256+1, 2048) f16, last row = pad
    const float*    __restrict__ bias,  // (256,)
    float*          __restrict__ y)     // (65536, 256) f32
{
    // tanh(x) tile, f16, 64 rows x 256 cols; within each 4-col group columns
    // are stored in natural order (d, d+1, d+2, d+3) so each lane's two A inputs for a
    // K-step are one aligned f16 pair (single ds_load_b32).
    __shared__ _Float16 Xs[64 * XPADH];

    const int tid      = threadIdx.x;
    const int lane     = tid & 31;
    const int wave     = tid >> 5;
    const int blockRow = blockIdx.x * 64;

    // ---- stage tanh(x) tile into LDS (coalesced float4 -> v4h, natural order) ----
    {
        const float4* xg = (const float4*)(x + (size_t)blockRow * DIM);
        #pragma unroll
        for (int it = 0; it < 16; ++it) {
            int idx = it * 256 + tid;          // 0 .. 4095 float4's
            int r   = idx >> 6;                // 64 float4 per row
            int c4  = idx & 63;                // one aligned group of 4 cols
            float4 v = xg[idx];
            v4h o = { (_Float16)fast_tanh(v.x), (_Float16)fast_tanh(v.y),
                      (_Float16)fast_tanh(v.z), (_Float16)fast_tanh(v.w) };
            *(v4h*)&Xs[r * XPADH + c4 * 4] = o;
        }
    }
    __syncthreads();

    const int rg     = wave >> 2;        // 0..1 : 32-row group
    const int nsub   = wave & 3;         // 0..3 : 64-col slice
    const int laneHi = lane >> 4;        // 0 | 1 (upper half-wave)
    const int ncol   = lane & 15;
    const int n0     = nsub * 64;
    const int rowA   = rg * 32 + ncol;   // A-frag-0 row this lane feeds
    const int rowB   = rowA + 16;        // A-frag-1 row

    const _Float16* xrA = &Xs[rowA * XPADH + laneHi * 2];
    const _Float16* xrB = &Xs[rowB * XPADH + laneHi * 2];

    // B fragment base: lane<16 reads K=[k0,k0+16), lane>=16 reads K=[k0+16,k0+32)
    const int koff = laneHi * 16;
    const _Float16* b0 = B + (size_t)(n0 +  0 + ncol) * RED + koff;
    const _Float16* b1 = B + (size_t)(n0 + 16 + ncol) * RED + koff;
    const _Float16* b2 = B + (size_t)(n0 + 32 + ncol) * RED + koff;
    const _Float16* b3 = B + (size_t)(n0 + 48 + ncol) * RED + koff;

    v8f accA0 = {}, accA1 = {}, accA2 = {}, accA3 = {};
    v8f accB0 = {}, accB1 = {}, accB2 = {}, accB3 = {};

    // prefetch B fragments for K-step 0
    v16h nb0 = *(const v16h*)(b0);
    v16h nb1 = *(const v16h*)(b1);
    v16h nb2 = *(const v16h*)(b2);
    v16h nb3 = *(const v16h*)(b3);

    for (int ks = 0; ks < 64; ++ks) {
        v16h bf0 = nb0, bf1 = nb1, bf2 = nb2, bf3 = nb3;

        // ---- prefetch next K-step's B fragments, unconditionally.
        // Pure induction addressing (immediate offsets, no wrap-select); the
        // final iteration reads 32 halfwords past each column's row, which the
        // one padding row at the end of the weight buffer keeps in-bounds.
        const int k1 = (ks + 1) * 32;
        nb0 = *(const v16h*)(b0 + k1);
        nb1 = *(const v16h*)(b1 + k1);
        nb2 = *(const v16h*)(b2 + k1);
        nb3 = *(const v16h*)(b3 + k1);

        // ---- build the two A fragments (16x32 f16 each) ----
        // K-chunk covers d = 4*ks .. 4*ks+3; row-major LDS layout makes this
        // lane's pair (d0, d0+1) adjacent -> one ds_load_b32 per fragment.
        v2h pa = *(const v2h*)(xrA + ks * 4);
        v2h pb = *(const v2h*)(xrB + ks * 4);

        v16h aA, aB;
        cheby8((float)pa[0], aA, 0);
        cheby8((float)pa[1], aA, 8);
        cheby8((float)pb[0], aB, 0);
        cheby8((float)pb[1], aB, 8);

        accA0 = __builtin_amdgcn_wmma_f32_16x16x32_f16(false, aA, false, bf0,
                                                       (short)0, accA0, false, false);
        accB0 = __builtin_amdgcn_wmma_f32_16x16x32_f16(false, aB, false, bf0,
                                                       (short)0, accB0, false, false);
        accA1 = __builtin_amdgcn_wmma_f32_16x16x32_f16(false, aA, false, bf1,
                                                       (short)0, accA1, false, false);
        accB1 = __builtin_amdgcn_wmma_f32_16x16x32_f16(false, aB, false, bf1,
                                                       (short)0, accB1, false, false);
        accA2 = __builtin_amdgcn_wmma_f32_16x16x32_f16(false, aA, false, bf2,
                                                       (short)0, accA2, false, false);
        accB2 = __builtin_amdgcn_wmma_f32_16x16x32_f16(false, aB, false, bf2,
                                                       (short)0, accB2, false, false);
        accA3 = __builtin_amdgcn_wmma_f32_16x16x32_f16(false, aA, false, bf3,
                                                       (short)0, accA3, false, false);
        accB3 = __builtin_amdgcn_wmma_f32_16x16x32_f16(false, aB, false, bf3,
                                                       (short)0, accB3, false, false);
    }

    // ---- epilogue: C layout (vgpr i: lanes 0-15 -> M=i, lanes 16-31 -> M=8+i) ----
    const int rowbaseA = blockRow + rg * 32 + laneHi * 8;
    const int rowbaseB = rowbaseA + 16;
    {
        int n = n0 + 0 + ncol;  float bz = bias[n];
        #pragma unroll
        for (int i = 0; i < 8; ++i) {
            y[(size_t)(rowbaseA + i) * OUTC + n] = accA0[i] + bz;
            y[(size_t)(rowbaseB + i) * OUTC + n] = accB0[i] + bz;
        }
    }
    {
        int n = n0 + 16 + ncol; float bz = bias[n];
        #pragma unroll
        for (int i = 0; i < 8; ++i) {
            y[(size_t)(rowbaseA + i) * OUTC + n] = accA1[i] + bz;
            y[(size_t)(rowbaseB + i) * OUTC + n] = accB1[i] + bz;
        }
    }
    {
        int n = n0 + 32 + ncol; float bz = bias[n];
        #pragma unroll
        for (int i = 0; i < 8; ++i) {
            y[(size_t)(rowbaseA + i) * OUTC + n] = accA2[i] + bz;
            y[(size_t)(rowbaseB + i) * OUTC + n] = accB2[i] + bz;
        }
    }
    {
        int n = n0 + 48 + ncol; float bz = bias[n];
        #pragma unroll
        for (int i = 0; i < 8; ++i) {
            y[(size_t)(rowbaseA + i) * OUTC + n] = accA3[i] + bz;
            y[(size_t)(rowbaseB + i) * OUTC + n] = accB3[i] + bz;
        }
    }
}

extern "C" void kernel_launch(void* const* d_in, const int* in_sizes, int n_in,
                              void* d_out, int out_size, void* d_ws, size_t ws_size,
                              hipStream_t stream) {
    const float* x      = (const float*)d_in[0];   // (65536, 256)
    const float* coeffs = (const float*)d_in[1];   // (256, 256, 8)
    const float* bias   = (const float*)d_in[2];   // (1, 256)
    float*       y      = (float*)d_out;           // (65536, 256)
    _Float16*    Bw     = (_Float16*)d_ws;         // 1 MB f16 weights + 4 KB pad row

    // 1) convert weights to f16 in workspace (2 MB read, 1 MB write).
    //    Workspace use: OUTC*RED*2 bytes + one 4 KB padding row for the
    //    unconditional last-iteration B prefetch.
    cvt_coeffs_f16<<<(OUTC * RED / 4) / 256, 256, 0, stream>>>(coeffs, Bw);

    // 2) fused tanh + Chebyshev + WMMA GEMM + bias
    kan_wmma_gemm<<<N_TOK / 64, 256, 0, stream>>>(x, Bw, bias, y);
}
